// ReZeroCorrectionSeq2Seq_44263932953250
// MI455X (gfx1250) — matrix operand order repaired and checked by reading, not compiled, not run on hardware
//
#include <hip/hip_runtime.h>

// Problem constants (reference: T=256, B=128, V=512, H=512)
#define T_ 256
#define B_ 128
#define V_ 512
#define H_ 512
#define TB_ (T_ * B_)

typedef __bf16 bf16;
typedef __attribute__((ext_vector_type(16))) __bf16 v16bf;
typedef __attribute__((ext_vector_type(8)))  float  v8f;
typedef __attribute__((ext_vector_type(4)))  unsigned int u32x4;

union V16U { v16bf v; u32x4 u[2]; };

// A-matrix 16x32 bf16 tile (M x K), per CDNA5 ISA layout:
// lane<16: row m0+lane, K = k+{0..7} then k+{16..23}
// lane>=16: row m0+(lane-16), K = k+{8..15} then k+{24..31}
__device__ inline v16bf load_a_tile(const bf16* __restrict__ A, int lda, int m0, int k, int lane) {
    int row  = m0 + (lane & 15);
    int koff = (lane < 16) ? 0 : 8;
    const bf16* p = A + (size_t)row * lda + k + koff;
    V16U r;
    r.u[0] = *(const u32x4*)(p);        // 8 bf16
    r.u[1] = *(const u32x4*)(p + 16);   // 8 bf16, +16 elements (upper K half)
    return r.v;
}

// B-matrix 32x16 bf16 tile (K x N) where B[k][n] = W[n0+n][k]:
// lanes 0-15 hold K = k+{0..7,16..23} for column n0+lane; lanes 16-31 hold K = k+{8..15,24..31}
__device__ inline v16bf load_b_tile(const bf16* __restrict__ W, int ldw, int n0, int k, int lane) {
    int row  = n0 + (lane & 15);
    int koff = (lane < 16) ? 0 : 8;
    const bf16* p = W + (size_t)row * ldw + k + koff;
    V16U r;
    r.u[0] = *(const u32x4*)(p);        // elems 0..7
    r.u[1] = *(const u32x4*)(p + 16);   // elems 8..15
    return r.v;
}

__device__ inline v8f wmma_bf16(v16bf a, v16bf b, v8f c) {
    return __builtin_amdgcn_wmma_f32_16x16x32_bf16(false, a, false, b, (short)0, c, false, false);
}

__device__ inline float sigmoidf_dev(float x) {
    return 1.0f / (1.0f + __expf(-x));
}

// ---------------------------------------------------------------------------
// Generic bf16 GEMM: C[M,N] = A[M,K] @ W[N,K]^T (+ biasN[n]) (+ biasRB[m%B][n])
// Output either bf16 (Ob) or f32 (Of, with optional accumulate).
// Block: 256 threads = 8 waves as 2(M) x 4(N); block tile 64M x 256N;
// each wave computes 32M x 64N via 8 accumulators.
// ---------------------------------------------------------------------------
__global__ __launch_bounds__(256) void gemm_bf16_kernel(
    const bf16* __restrict__ A, int lda,
    const bf16* __restrict__ W, int ldw,
    int K,
    const float* __restrict__ biasN,
    const float* __restrict__ biasRB, int ldrb,
    bf16* __restrict__ Ob, float* __restrict__ Of, int ldc,
    int accumulate)
{
    const int lane = threadIdx.x & 31;
    const int wv   = threadIdx.x >> 5;
    const int m0 = blockIdx.y * 64 + (wv >> 2) * 32;
    const int n0 = blockIdx.x * 256 + (wv & 3) * 64;

    v8f acc[2][4] = {};

    for (int k = 0; k < K; k += 32) {
        v16bf a0 = load_a_tile(A, lda, m0,      k, lane);
        v16bf a1 = load_a_tile(A, lda, m0 + 16, k, lane);
        if (k + 32 < K) {
            __builtin_prefetch(A + (size_t)(m0 + (lane & 15)) * lda + k + 32, 0, 3);
        }
#pragma unroll
        for (int nt = 0; nt < 4; ++nt) {
            v16bf b = load_b_tile(W, ldw, n0 + nt * 16, k, lane);
            acc[0][nt] = wmma_bf16(a0, b, acc[0][nt]);
            acc[1][nt] = wmma_bf16(a1, b, acc[1][nt]);
        }
    }

    const int mrow_off = (lane >> 4) << 3;   // lane>=16 -> rows +8
    const int ncol     = lane & 15;
#pragma unroll
    for (int mt = 0; mt < 2; ++mt)
#pragma unroll
    for (int nt = 0; nt < 4; ++nt)
#pragma unroll
    for (int r = 0; r < 8; ++r) {
        int m = m0 + mt * 16 + mrow_off + r;
        int n = n0 + nt * 16 + ncol;
        float v = acc[mt][nt][r];
        if (biasN)  v += biasN[n];
        if (biasRB) v += biasRB[(size_t)(m & (B_ - 1)) * ldrb + n];
        size_t idx = (size_t)m * ldc + n;
        if (Ob)              Ob[idx] = (bf16)v;
        else if (accumulate) Of[idx] += v;
        else                 Of[idx] = v;
    }
}

// ---------------------------------------------------------------------------
// One GRU time step for BOTH encoder directions, fused:
//   gh = h @ Whh^T (WMMA, K=512, three gate blocks), then GRU update,
//   running ctx sum accumulation, double-buffered h (f32 + bf16 copies).
// Grid: 16 blocks (2 dirs x 8 column-blocks of 64), 256 threads (8 waves,
// wave w owns batch rows 16w..16w+15). Batch = GEMM M dimension (128).
// ---------------------------------------------------------------------------
__global__ __launch_bounds__(256) void rec_step_kernel(
    const bf16* __restrict__ gxf, const bf16* __restrict__ gxb,
    const bf16* __restrict__ whhf, const bf16* __restrict__ whhb,
    const float* __restrict__ bhhf, const float* __restrict__ bhhb,
    float* __restrict__ h32, bf16* __restrict__ h16,
    float* __restrict__ ctxs, int t)
{
    const int lane = threadIdx.x & 31;
    const int wv   = threadIdx.x >> 5;
    const int dir  = blockIdx.x >> 3;
    const int j0   = (blockIdx.x & 7) * 64;
    const int m0   = wv * 16;

    const bf16*  whh = dir ? whhb : whhf;
    const float* bhh = dir ? bhhb : bhhf;
    const bf16*  gx  = dir ? gxb  : gxf;
    const int trow = dir ? (T_ - 1 - t) : t;
    const int cur = t & 1, nxt = cur ^ 1;

    const bf16* hA = h16 + (size_t)(dir * 2 + cur) * (B_ * H_);

    v8f acc[3][4] = {};
    for (int k = 0; k < H_; k += 32) {
        v16bf a = load_a_tile(hA, H_, m0, k, lane);
#pragma unroll
        for (int g = 0; g < 3; ++g)
#pragma unroll
        for (int nt = 0; nt < 4; ++nt) {
            v16bf b = load_b_tile(whh, H_, g * H_ + j0 + nt * 16, k, lane);
            acc[g][nt] = wmma_bf16(a, b, acc[g][nt]);
        }
    }

    const int mrow_off = (lane >> 4) << 3;
    const int ncol     = lane & 15;
    const float* hprev32 = h32 + (size_t)(dir * 2 + cur) * (B_ * H_);
    float*       hnext32 = h32 + (size_t)(dir * 2 + nxt) * (B_ * H_);
    bf16*        hnext16 = h16 + (size_t)(dir * 2 + nxt) * (B_ * H_);

#pragma unroll
    for (int nt = 0; nt < 4; ++nt)
#pragma unroll
    for (int r = 0; r < 8; ++r) {
        int m = m0 + mrow_off + r;      // batch index
        int j = j0 + nt * 16 + ncol;    // hidden index
        float ghr = acc[0][nt][r] + bhh[j];
        float ghz = acc[1][nt][r] + bhh[H_ + j];
        float ghn = acc[2][nt][r] + bhh[2 * H_ + j];
        size_t gidx = ((size_t)trow * B_ + m) * (3 * H_);
        float gxr = (float)gx[gidx + j];
        float gxz = (float)gx[gidx + H_ + j];
        float gxn = (float)gx[gidx + 2 * H_ + j];
        float rg = sigmoidf_dev(gxr + ghr);
        float zg = sigmoidf_dev(gxz + ghz);
        float ng = tanhf(gxn + rg * ghn);
        float hp = hprev32[(size_t)m * H_ + j];
        float h2 = (1.0f - zg) * ng + zg * hp;
        hnext32[(size_t)m * H_ + j] = h2;
        hnext16[(size_t)m * H_ + j] = (bf16)h2;
        ctxs[(size_t)m * (2 * H_) + dir * H_ + j] += h2;   // ctx = sum_t enc_out
    }
}

// ---------------------------------------------------------------------------
// argmax over V=512 per (t,b) row; one wave per row, 8 rows per block.
// ---------------------------------------------------------------------------
__global__ __launch_bounds__(256) void argmax_kernel(const float* __restrict__ x,
                                                     int* __restrict__ ids)
{
    int lane = threadIdx.x & 31;
    int wv   = threadIdx.x >> 5;
    int row  = blockIdx.x * 8 + wv;
    const float* p = x + (size_t)row * V_;
    float best = -3.0e38f; int bi = 0;
    for (int v = lane; v < V_; v += 32) {
        float f = p[v];
        if (f > best) { best = f; bi = v; }
    }
    for (int off = 16; off > 0; off >>= 1) {
        float ob = __shfl_down(best, off, 32);
        int   oi = __shfl_down(bi,   off, 32);
        if (ob > best || (ob == best && oi < bi)) { best = ob; bi = oi; }
    }
    if (lane == 0) ids[row] = bi;
}

// Gather both embedding tables for a row, converting f32 -> bf16.
__global__ __launch_bounds__(256) void gather_emb_kernel(
    const int* __restrict__ ids,
    const float* __restrict__ enc_emb, const float* __restrict__ dec_emb,
    bf16* __restrict__ embK, bf16* __restrict__ demK)
{
    int row = blockIdx.x;
    int id  = ids[row];
    const float* e = enc_emb + (size_t)id * H_;
    const float* d = dec_emb + (size_t)id * H_;
    bf16* eo = embK + (size_t)row * H_;
    bf16* oo = demK + (size_t)row * H_;
    for (int j = threadIdx.x; j < H_; j += 256) {
        eo[j] = (bf16)e[j];
        oo[j] = (bf16)d[j];
    }
}

__global__ void cvt_f32_bf16_kernel(const float* __restrict__ s, bf16* __restrict__ d, int n) {
    int i = blockIdx.x * blockDim.x + threadIdx.x;
    if (i < n) d[i] = (bf16)s[i];
}

// Decoder gates with h0=0: hstep = (1-z)*n, gx = gxd(row) + ctxg(b).
__global__ __launch_bounds__(256) void dec_gate_kernel(
    const bf16* __restrict__ gxd, const float* __restrict__ ctxg,
    const float* __restrict__ bhh, bf16* __restrict__ hstep)
{
    int row = blockIdx.x;
    int b = row & (B_ - 1);
    const bf16*  g = gxd  + (size_t)row * (3 * H_);
    const float* c = ctxg + (size_t)b   * (3 * H_);
    for (int j = threadIdx.x; j < H_; j += 256) {
        float xr = (float)g[j]            + c[j];
        float xz = (float)g[H_ + j]       + c[H_ + j];
        float xn = (float)g[2 * H_ + j]   + c[2 * H_ + j];
        float rg = sigmoidf_dev(xr + bhh[j]);
        float zg = sigmoidf_dev(xz + bhh[H_ + j]);
        float ng = tanhf(xn + rg * bhh[2 * H_ + j]);
        hstep[(size_t)row * H_ + j] = (bf16)((1.0f - zg) * ng);
    }
}

// Final ReZero blend + [T,B,V] -> [B,T,V] transpose.
__global__ __launch_bounds__(256) void blend_kernel(
    const float* __restrict__ ob, const float* __restrict__ x,
    const float* __restrict__ alpha, float* __restrict__ out)
{
    size_t i = (size_t)blockIdx.x * blockDim.x + threadIdx.x;   // over B*T*V
    int v = (int)(i & (V_ - 1));
    int t = (int)((i >> 9) & (T_ - 1));
    int b = (int)(i >> 17);
    float a = sigmoidf_dev(alpha[0]);    // alpha = -1e12 -> a == 0
    size_t src = ((size_t)t * B_ + b) * V_ + v;
    out[i] = ob[src] * a + x[src] * (1.0f - a);
}

// ---------------------------------------------------------------------------
extern "C" void kernel_launch(void* const* d_in, const int* in_sizes, int n_in,
                              void* d_out, int out_size, void* d_ws, size_t ws_size,
                              hipStream_t stream)
{
    (void)in_sizes; (void)n_in; (void)out_size; (void)ws_size;

    const float* x         = (const float*)d_in[0];
    const float* enc_emb   = (const float*)d_in[1];
    const float* enc_Wih_f = (const float*)d_in[2];
    const float* enc_Whh_f = (const float*)d_in[3];
    const float* enc_bih_f = (const float*)d_in[4];
    const float* enc_bhh_f = (const float*)d_in[5];
    const float* enc_Wih_b = (const float*)d_in[6];
    const float* enc_Whh_b = (const float*)d_in[7];
    const float* enc_bih_b = (const float*)d_in[8];
    const float* enc_bhh_b = (const float*)d_in[9];
    // d_in[10..11] enc_fcW/fcb, d_in[13..15] attn*, d_in[17] dec_Whh: dead code.
    const float* dec_emb   = (const float*)d_in[12];
    const float* dec_Wih   = (const float*)d_in[16];
    const float* dec_bih   = (const float*)d_in[18];
    const float* dec_bhh   = (const float*)d_in[19];
    const float* fcW       = (const float*)d_in[20];
    const float* fcb       = (const float*)d_in[21];
    const float* alpha     = (const float*)d_in[22];

    // ---- workspace layout ----
    char* ws = (char*)d_ws;
    size_t off = 0;
    auto alloc = [&](size_t bytes) -> void* {
        off = (off + 255) & ~(size_t)255;
        void* p = ws + off;
        off += bytes;
        return p;
    };
    int*  ids   = (int*) alloc((size_t)TB_ * 4);
    bf16* embK  = (bf16*)alloc((size_t)TB_ * H_ * 2);
    bf16* demK  = (bf16*)alloc((size_t)TB_ * H_ * 2);
    bf16* w_ihf = (bf16*)alloc((size_t)3 * H_ * H_ * 2);
    bf16* w_hhf = (bf16*)alloc((size_t)3 * H_ * H_ * 2);
    bf16* w_ihb = (bf16*)alloc((size_t)3 * H_ * H_ * 2);
    bf16* w_hhb = (bf16*)alloc((size_t)3 * H_ * H_ * 2);
    bf16* w_dec = (bf16*)alloc((size_t)3 * H_ * 3 * H_ * 2);
    bf16* w_fc  = (bf16*)alloc((size_t)V_ * 4 * H_ * 2);
    bf16* gxf   = (bf16*)alloc((size_t)TB_ * 3 * H_ * 2);   // later reused for gxd
    bf16* gxb   = (bf16*)alloc((size_t)TB_ * 3 * H_ * 2);   // later reused as f32 outbuf
    float* h32  = (float*)alloc((size_t)4 * B_ * H_ * 4);   // 2 dirs x 2 bufs
    bf16*  h16  = (bf16*) alloc((size_t)4 * B_ * H_ * 2);
    float* ctxs = (float*)alloc((size_t)B_ * 2 * H_ * 4);
    bf16*  ctxbf= (bf16*) alloc((size_t)B_ * 2 * H_ * 2);
    float* ctxg = (float*)alloc((size_t)B_ * 3 * H_ * 4);
    float* ctxfc= (float*)alloc((size_t)B_ * V_ * 4);
    bf16* hstep = (bf16*)alloc((size_t)TB_ * H_ * 2);

    auto gemm = [&](const bf16* A, int lda, const bf16* W, int ldw,
                    int M, int N, int K,
                    const float* biasN, const float* biasRB, int ldrb,
                    bf16* Ob, float* Of, int ldc, int acc) {
        dim3 g(N / 256, M / 64);
        gemm_bf16_kernel<<<g, 256, 0, stream>>>(A, lda, W, ldw, K, biasN,
                                                biasRB, ldrb, Ob, Of, ldc, acc);
    };
    auto cvt = [&](const float* s, bf16* d, int n) {
        cvt_f32_bf16_kernel<<<(n + 255) / 256, 256, 0, stream>>>(s, d, n);
    };

    // 0) zero recurrence state + ctx accumulator
    hipMemsetAsync(h32,  0, (size_t)4 * B_ * H_ * 4, stream);
    hipMemsetAsync(h16,  0, (size_t)4 * B_ * H_ * 2, stream);
    hipMemsetAsync(ctxs, 0, (size_t)B_ * 2 * H_ * 4, stream);

    // 1) weights -> bf16
    cvt(enc_Wih_f, w_ihf, 3 * H_ * H_);
    cvt(enc_Whh_f, w_hhf, 3 * H_ * H_);
    cvt(enc_Wih_b, w_ihb, 3 * H_ * H_);
    cvt(enc_Whh_b, w_hhb, 3 * H_ * H_);
    cvt(dec_Wih,   w_dec, 3 * H_ * 3 * H_);
    cvt(fcW,       w_fc,  V_ * 4 * H_);

    // 2) ids = argmax(x, -1); gather embeddings (bf16)
    argmax_kernel<<<TB_ / 8, 256, 0, stream>>>(x, ids);
    gather_emb_kernel<<<TB_, 256, 0, stream>>>(ids, enc_emb, dec_emb, embK, demK);

    // 3) precompute encoder input gates gx = emb @ Wih^T + bih (both dirs)
    gemm(embK, H_, w_ihf, H_, TB_, 3 * H_, H_, enc_bih_f, nullptr, 0, gxf, nullptr, 3 * H_, 0);
    gemm(embK, H_, w_ihb, H_, TB_, 3 * H_, H_, enc_bih_b, nullptr, 0, gxb, nullptr, 3 * H_, 0);

    // 4) sequential bidirectional GRU recurrence (fused fwd+bwd per step)
    for (int t = 0; t < T_; ++t) {
        rec_step_kernel<<<16, 256, 0, stream>>>(gxf, gxb, w_hhf, w_hhb,
                                                enc_bhh_f, enc_bhh_b,
                                                h32, h16, ctxs, t);
    }

    // 5) ctx (= sum_t enc_out) -> bf16; per-batch projections
    cvt(ctxs, ctxbf, B_ * 2 * H_);
    // ctxg = ctx @ dec_Wih[:, H:3H]^T + dec_bih           [B, 3H]
    gemm(ctxbf, 2 * H_, w_dec + H_, 3 * H_, B_, 3 * H_, 2 * H_,
         dec_bih, nullptr, 0, nullptr, ctxg, 3 * H_, 0);
    // ctxfc = ctx @ fcW[:, H:3H]^T + fcb                  [B, V]
    gemm(ctxbf, 2 * H_, w_fc + H_, 4 * H_, B_, V_, 2 * H_,
         fcb, nullptr, 0, nullptr, ctxfc, V_, 0);

    // 6) decoder input gates: gxd = dem @ dec_Wih[:, 0:H]^T   (reuse gxf buffer)
    gemm(demK, H_, w_dec, 3 * H_, TB_, 3 * H_, H_,
         nullptr, nullptr, 0, gxf, nullptr, 3 * H_, 0);

    // 7) decoder GRU gates with h0=0 -> hstep (bf16)
    dec_gate_kernel<<<TB_, 256, 0, stream>>>(gxf, ctxg, dec_bhh, hstep);

    // 8) output head: out = hstep@fcW[:,0:H]^T + ctxfc[b] ; out += dem@fcW[:,3H:4H]^T
    float* outbuf = (float*)gxb;   // reuse (100 MB region, need 64 MB f32)
    gemm(hstep, H_, w_fc, 4 * H_, TB_, V_, H_,
         nullptr, ctxfc, V_, nullptr, outbuf, V_, 0);
    gemm(demK, H_, w_fc + 3 * H_, 4 * H_, TB_, V_, H_,
         nullptr, nullptr, 0, nullptr, outbuf, V_, 1);

    // 9) ReZero blend with x and transpose to [B,T,V]
    blend_kernel<<<(B_ * T_ * V_) / 256, 256, 0, stream>>>(outbuf, x, alpha, (float*)d_out);
}
